// Model_17944373363287
// MI455X (gfx1250) — matrix operand order repaired and checked by reading, not compiled, not run on hardware
//
#include <hip/hip_runtime.h>
#include <hip/hip_bf16.h>
#include <math.h>

// ---------------------------------------------------------------------------
// Types for CDNA5 WMMA / TDM
// ---------------------------------------------------------------------------
typedef __attribute__((ext_vector_type(16))) _Float16 v16h;
typedef __attribute__((ext_vector_type(8)))  float    v8f;
typedef __attribute__((ext_vector_type(4)))  unsigned u32x4;
typedef __attribute__((ext_vector_type(8)))  unsigned u32x8;

#define SEQ 2048
#define E_DIM 300
#define H_DIM 150
#define G_DIM 600      // 4*H
#define D_DIM 300      // 2*H
#define KP    320      // K padded to multiple of 32 (300 -> 320)
#define NP1   640      // 600 -> padded to multiple of 128
#define NP2   384      // 300 -> padded to multiple of 128

// ---------------------------------------------------------------------------
// TDM: 2D tile load Global -> LDS via Tensor Data Mover (inline asm form,
// portable across toolchains). Descriptor per CDNA5 ISA.
// data_size = 2 bytes (f16). Issued wave-uniform.
// ---------------------------------------------------------------------------
__device__ __forceinline__ void tdm_load_2d_f16(unsigned lds_addr,
                                                const _Float16* gptr,
                                                unsigned tensor_d0,   // elems
                                                unsigned tile_d0,     // elems
                                                unsigned tile_d1,     // rows
                                                unsigned stride0)     // elems
{
    unsigned long long ga = (unsigned long long)(const void*)gptr;
    const unsigned td1_hi = 0x10u;             // tensor_dim1 = 1<<20 (no OOB)
    u32x4 g0 = {
        1u,                                    // count=1, user mode
        lds_addr,                              // lds_addr (bytes)
        (unsigned)ga,                          // global_addr[31:0]
        (unsigned)((ga >> 32) & 0x1FFFFFFull) | 0x80000000u  // [56:32]|type=2
    };
    u32x8 g1 = {
        (1u << 16),                                    // data_size = 2 bytes
        (tensor_d0 & 0xFFFFu) << 16,                   // tensor_dim0[15:0]
        (tensor_d0 >> 16),                             // tensor_dim0[31:16], td1 lo=0
        td1_hi | (tile_d0 << 16),                      // td1 hi | tile_dim0
        tile_d1 & 0xFFFFu,                             // tile_dim1, tile_dim2=0
        stride0,                                       // tensor_dim0_stride[31:0]
        0u,                                            // stride0[47:32], stride1 lo
        0u                                             // stride1 hi
    };
    u32x4 gz = { 0u, 0u, 0u, 0u };                     // groups 2/3 unused (2D)
    asm volatile("tensor_load_to_lds %0, %1, %2, %3"
                 :: "s"(g0), "s"(g1), "s"(gz), "s"(gz)
                 : "memory");
}

// ---------------------------------------------------------------------------
// K1: gather embeddings -> f16 (K-padded to 320, pad cols zeroed)
// ---------------------------------------------------------------------------
__global__ void gather_embed(const int* __restrict__ sentence,
                             const float* __restrict__ emb_mat,
                             _Float16* __restrict__ emb16)
{
    int s = blockIdx.x;
    int e = threadIdx.x;                       // 0..319
    int tok = sentence[s];
    float v = (e < E_DIM) ? emb_mat[(size_t)tok * E_DIM + e] : 0.0f;
    emb16[(size_t)s * KP + e] = (_Float16)v;
}

// ---------------------------------------------------------------------------
// K2: weight prep: W [K][N] f32 -> Wt [Np][Kp] f16 (B^T, zero-padded)
// ---------------------------------------------------------------------------
__global__ void prep_wt(const float* __restrict__ W, _Float16* __restrict__ Wt,
                        int K, int N, int Kp, int Np)
{
    int idx = blockIdx.x * blockDim.x + threadIdx.x;
    if (idx >= Kp * Np) return;
    int n = idx / Kp;
    int k = idx - n * Kp;
    float v = (n < N && k < K) ? W[(size_t)k * N + n] : 0.0f;
    Wt[idx] = (_Float16)v;
}

// ---------------------------------------------------------------------------
// K3/K5: WMMA GEMM with TDM->LDS staging.
//   C[M][N] = A[M][Kp] * Bt[Np][Kp]^T + bias[N]
// Block = 256 thr (8 waves). Block covers one 16-row M tile x 128 columns.
// Wave 0 stages A tile (16xKp) and a 128-row Bt slab into LDS via TDM,
// waits TENSORcnt, barrier; then each wave runs v_wmma over K from LDS.
// ---------------------------------------------------------------------------
__global__ void __launch_bounds__(256)
wmma_gemm_tdm(const _Float16* __restrict__ A, const _Float16* __restrict__ Bt,
              const float* __restrict__ bias, float* __restrict__ C,
              int M, int N, int Np, int Kp, int ldc)
{
    extern __shared__ char smem[];
    _Float16* Ash = (_Float16*)smem;                   // 16  x Kp
    _Float16* Bsh = (_Float16*)(smem + 16 * Kp * 2);   // 128 x Kp

    const int nbBlk = Np >> 7;                 // 128-column groups
    const int tm = blockIdx.x / nbBlk;         // M tile (16 rows)
    const int tb = blockIdx.x - tm * nbBlk;    // which 128-column group
    const int wId  = threadIdx.x >> 5;         // 0..7 -> N sub-tile
    const int lane = threadIdx.x & 31;
    const int r    = lane & 15;
    const int hi   = lane >> 4;

    if (threadIdx.x < 32) {                    // wave 0 stages via TDM
        unsigned ldsA = (unsigned)(unsigned long long)(void*)Ash;
        unsigned ldsB = (unsigned)(unsigned long long)(void*)Bsh;
        tdm_load_2d_f16(ldsA, A  + (size_t)tm * 16  * Kp, Kp, Kp, 16,  Kp);
        tdm_load_2d_f16(ldsB, Bt + (size_t)tb * 128 * Kp, Kp, Kp, 128, Kp);
        __builtin_amdgcn_s_wait_tensorcnt(0);
    }
    __syncthreads();

    // LDS fragment base pointers (16B-aligned rows; Kp multiple of 32)
    const uint4* ArowQ = (const uint4*)(Ash + (size_t)r * Kp);
    const uint4* BrowQ = (const uint4*)(Bsh + (size_t)(wId * 16 + r) * Kp);

    v8f acc = {};
    for (int k0 = 0; k0 < Kp; k0 += 32) {
        union { v16h h; uint4 q[2]; } a, b;
        int qk = k0 >> 3;                      // uint4 index of K chunk
        // A 16x32 f16 layout: lanes 0-15 K=0..7 / 16..23, lanes 16-31 +8
        a.q[0] = ArowQ[qk + hi];
        a.q[1] = ArowQ[qk + hi + 2];
        // B 32x16 f16 layout: lanes 0-15 K=0..7 / 16..23, lanes 16-31 +8
        b.q[0] = BrowQ[qk + hi];
        b.q[1] = BrowQ[qk + hi + 2];

        acc = __builtin_amdgcn_wmma_f32_16x16x32_f16(
            /*neg_a=*/false, a.h, /*neg_b=*/false, b.h,
            /*c_mod=*/(short)0, acc, /*reuse_a=*/false, /*reuse_b=*/false);
    }

    int n = (tb * 8 + wId) * 16 + r;
    if (n < N) {
        float bb = bias ? bias[n] : 0.0f;
#pragma unroll
        for (int i = 0; i < 8; ++i) {
            int m = tm * 16 + i + hi * 8;      // C layout: VGPR i -> M=i / M=i+8
            C[(size_t)m * ldc + n] = acc[i] + bb;
        }
    }
}

// ---------------------------------------------------------------------------
// K4: LSTM scan, both directions (blockIdx.x: 0=fwd, 1=bwd).
// U (150x600) resident in LDS as f16 (180 KB of the 320 KB WGP LDS).
// 640 threads = 20 wave32; threads 0..599 each own one gate column.
// Prefetch next step's xp row while computing the current matvec.
// ---------------------------------------------------------------------------
__global__ void __launch_bounds__(640)
lstm_scan(const float* __restrict__ xp_f, const float* __restrict__ xp_b,
          const float* __restrict__ Uf,   const float* __restrict__ Ub,
          float* __restrict__ hidden, _Float16* __restrict__ hidden16)
{
    extern __shared__ char smem[];
    _Float16* Ush = (_Float16*)smem;                       // 150*600 halves
    float*    zsh = (float*)(smem + H_DIM * G_DIM * 2);    // 600
    float*    hsh = zsh + G_DIM;                           // 152

    const int rev = blockIdx.x;
    const float* xp = rev ? xp_b : xp_f;
    const float* U  = rev ? Ub   : Uf;
    const int tid = threadIdx.x;

    for (int i = tid; i < H_DIM * G_DIM; i += 640) Ush[i] = (_Float16)U[i];
    if (tid < 152) hsh[tid] = 0.0f;
    float c = 0.0f;
    __syncthreads();

    for (int step = 0; step < SEQ; ++step) {
        int t = rev ? (SEQ - 1 - step) : step;
        if (tid < G_DIM) {
            // hide next step's xp fetch behind the LDS matvec
            int tn = rev ? (t - 1) : (t + 1);
            if (step + 1 < SEQ)
                __builtin_prefetch(&xp[(size_t)tn * G_DIM + tid], 0, 0);
            float acc = xp[(size_t)t * G_DIM + tid];
#pragma unroll 5
            for (int i = 0; i < H_DIM; ++i)
                acc += hsh[i] * (float)Ush[i * G_DIM + tid];
            zsh[tid] = acc;
        }
        __syncthreads();
        if (tid < H_DIM) {
            float zi = zsh[tid], zf = zsh[H_DIM + tid];
            float zg = zsh[2 * H_DIM + tid], zo = zsh[3 * H_DIM + tid];
            float ig = 1.0f / (1.0f + __expf(-zi));
            float fg = 1.0f / (1.0f + __expf(-zf));
            float g  = tanhf(zg);
            float og = 1.0f / (1.0f + __expf(-zo));
            c = fg * c + ig * g;
            float h = og * tanhf(c);
            hsh[tid] = h;
            int col = rev ? (H_DIM + tid) : tid;
            hidden[(size_t)t * D_DIM + col] = h;
            hidden16[(size_t)t * KP + col] = (_Float16)h;
        }
        __syncthreads();
    }
}

// ---------------------------------------------------------------------------
// K6: primary attention + secondary-attention coefficient, one block per s.
// ---------------------------------------------------------------------------
__global__ void __launch_bounds__(256)
attention(const float* __restrict__ proj,    // [S][300] projected hidden
          const float* __restrict__ hidden,  // [S][300]
          const int* __restrict__ sentence,
          const int* __restrict__ syn_idx,   // [VOCAB][4]
          const float* __restrict__ emb_mat,
          const float* __restrict__ w_sec,   // [600]
          const float* __restrict__ b_sec,   // [1]
          float* __restrict__ h_cat,         // [S][600]
          float* __restrict__ coeff)         // [S]
{
    __shared__ float synsh[4 * E_DIM];
    __shared__ float msh[E_DIM];
    __shared__ float red[256];
    __shared__ float att[4];
    __shared__ int   sidx[4];

    int s = blockIdx.x, tid = threadIdx.x;
    if (tid < 4) sidx[tid] = syn_idx[(size_t)sentence[s] * 4 + tid];
    __syncthreads();
    for (int j = tid; j < 4 * E_DIM; j += 256) {
        int k = j / E_DIM, e = j - k * E_DIM;
        synsh[j] = emb_mat[(size_t)sidx[k] * E_DIM + e];
    }
    __syncthreads();

    // att[k] = exp( proj[s] . syn[k] )
    float p[4] = {0, 0, 0, 0};
    for (int e = tid; e < E_DIM; e += 256) {
        float o = proj[(size_t)s * E_DIM + e];
#pragma unroll
        for (int k = 0; k < 4; ++k) p[k] += o * synsh[k * E_DIM + e];
    }
#pragma unroll
    for (int k = 0; k < 4; ++k) {
        red[tid] = p[k];
        __syncthreads();
        for (int w = 128; w > 0; w >>= 1) {
            if (tid < w) red[tid] += red[tid + w];
            __syncthreads();
        }
        if (tid == 0) att[k] = __expf(red[0]);
        __syncthreads();
    }

    // m[e] = sum_k att[k]*syn[k][e]; h_cat = [hidden | m]
    for (int e = tid; e < E_DIM; e += 256) {
        float m = att[0] * synsh[e] + att[1] * synsh[E_DIM + e] +
                  att[2] * synsh[2 * E_DIM + e] + att[3] * synsh[3 * E_DIM + e];
        msh[e] = m;
        h_cat[(size_t)s * G_DIM + E_DIM + e] = m;
        h_cat[(size_t)s * G_DIM + e] = hidden[(size_t)s * D_DIM + e];
    }
    __syncthreads();

    // coeff[s] = exp(tanh(h_cat[s] . w_sec + b))
    float q = 0.0f;
    for (int e = tid; e < G_DIM; e += 256) {
        float hv = (e < E_DIM) ? hidden[(size_t)s * D_DIM + e] : msh[e - E_DIM];
        q += hv * w_sec[e];
    }
    red[tid] = q;
    __syncthreads();
    for (int w = 128; w > 0; w >>= 1) {
        if (tid < w) red[tid] += red[tid + w];
        __syncthreads();
    }
    if (tid == 0) coeff[s] = __expf(tanhf(red[0] + b_sec[0]));
}

// ---------------------------------------------------------------------------
// K7: H[e] = sum_s coeff[s] * h_cat[s][e]
// ---------------------------------------------------------------------------
__global__ void weighted_sum(const float* __restrict__ h_cat,
                             const float* __restrict__ coeff,
                             float* __restrict__ Hout)
{
    int e = blockIdx.x * blockDim.x + threadIdx.x;
    if (e >= G_DIM) return;
    float acc = 0.0f;
    for (int s = 0; s < SEQ; ++s) acc += coeff[s] * h_cat[(size_t)s * G_DIM + e];
    Hout[e] = acc;
}

// ---------------------------------------------------------------------------
// K8: final logits. out[0..7] = emotion (H_BAR @ W_eo + b_eo),
//                   out[8]    = sentiment (H_HAT @ W_so + b_so)
// ---------------------------------------------------------------------------
__global__ void final_out(const float* __restrict__ H_HAT,
                          const float* __restrict__ H_BAR,
                          const float* __restrict__ W_eo, const float* __restrict__ b_eo,
                          const float* __restrict__ W_so, const float* __restrict__ b_so,
                          float* __restrict__ out)
{
    int j = threadIdx.x;
    if (j < 8) {
        float acc = b_eo[j];
        for (int e = 0; e < G_DIM; ++e) acc += H_BAR[e] * W_eo[e * 8 + j];
        out[j] = acc;
    } else if (j == 8) {
        float acc = b_so[0];
        for (int e = 0; e < G_DIM; ++e) acc += H_HAT[e] * W_so[e];
        out[8] = acc;
    }
}

// ---------------------------------------------------------------------------
// Host-side orchestration
// ---------------------------------------------------------------------------
extern "C" void kernel_launch(void* const* d_in, const int* in_sizes, int n_in,
                              void* d_out, int out_size, void* d_ws, size_t ws_size,
                              hipStream_t stream)
{
    (void)in_sizes; (void)n_in; (void)out_size; (void)ws_size;

    const int*   sentence = (const int*)  d_in[0];
    const float* emb_mat  = (const float*)d_in[1];
    const int*   syn_idx  = (const int*)  d_in[2];
    const float* Wf   = (const float*)d_in[3];
    const float* Uf   = (const float*)d_in[4];
    const float* bf   = (const float*)d_in[5];
    const float* Wb   = (const float*)d_in[6];
    const float* Ub   = (const float*)d_in[7];
    const float* bb   = (const float*)d_in[8];
    const float* W_pe = (const float*)d_in[9];
    const float* b_pe = (const float*)d_in[10];
    const float* W_ps = (const float*)d_in[11];
    const float* b_ps = (const float*)d_in[12];
    const float* w_se = (const float*)d_in[13];
    const float* b_se = (const float*)d_in[14];
    const float* w_ss = (const float*)d_in[15];
    const float* b_ss = (const float*)d_in[16];
    const float* W_eo = (const float*)d_in[17];
    const float* b_eo = (const float*)d_in[18];
    const float* W_so = (const float*)d_in[19];
    const float* b_so = (const float*)d_in[20];

    char* ws = (char*)d_ws;
    size_t off = 0;
    auto alloc = [&](size_t bytes) -> char* {
        char* p = ws + off;
        off = (off + bytes + 255) & ~(size_t)255;
        return p;
    };

    _Float16* emb16    = (_Float16*)alloc((size_t)SEQ * KP * 2);
    _Float16* WfT      = (_Float16*)alloc((size_t)NP1 * KP * 2);
    _Float16* WbT      = (_Float16*)alloc((size_t)NP1 * KP * 2);
    _Float16* WpsT     = (_Float16*)alloc((size_t)NP2 * KP * 2);
    _Float16* WpeT     = (_Float16*)alloc((size_t)NP2 * KP * 2);
    float*    xpf      = (float*)   alloc((size_t)SEQ * G_DIM * 4);
    float*    xpb      = (float*)   alloc((size_t)SEQ * G_DIM * 4);
    float*    hidden   = (float*)   alloc((size_t)SEQ * D_DIM * 4);
    _Float16* hidden16 = (_Float16*)alloc((size_t)SEQ * KP * 2);
    float*    out_s    = (float*)   alloc((size_t)SEQ * D_DIM * 4);
    float*    out_e    = (float*)   alloc((size_t)SEQ * D_DIM * 4);
    float*    h_hats   = (float*)   alloc((size_t)SEQ * G_DIM * 4);
    float*    h_bars   = (float*)   alloc((size_t)SEQ * G_DIM * 4);
    float*    coeff_s  = (float*)   alloc((size_t)SEQ * 4);
    float*    coeff_e  = (float*)   alloc((size_t)SEQ * 4);
    float*    H_HAT    = (float*)   alloc((size_t)G_DIM * 4);
    float*    H_BAR    = (float*)   alloc((size_t)G_DIM * 4);

    // zero-pad hidden16 (cols 300..319 must be 0 for the projection GEMM)
    hipMemsetAsync(hidden16, 0, (size_t)SEQ * KP * 2, stream);

    // K1: embedding gather -> f16
    gather_embed<<<SEQ, KP, 0, stream>>>(sentence, emb_mat, emb16);

    // K2: weight transpose + f16 convert (row counts padded to mult. of 128)
    {
        int n1 = NP1 * KP, n2 = NP2 * KP;
        prep_wt<<<(n1 + 255) / 256, 256, 0, stream>>>(Wf,   WfT,  E_DIM, G_DIM, KP, NP1);
        prep_wt<<<(n1 + 255) / 256, 256, 0, stream>>>(Wb,   WbT,  E_DIM, G_DIM, KP, NP1);
        prep_wt<<<(n2 + 255) / 256, 256, 0, stream>>>(W_ps, WpsT, D_DIM, D_DIM, KP, NP2);
        prep_wt<<<(n2 + 255) / 256, 256, 0, stream>>>(W_pe, WpeT, D_DIM, D_DIM, KP, NP2);
    }

    size_t gemm_smem = (size_t)(16 + 128) * KP * 2;   // 92160 B

    // K3: xp = emb @ W + b for both directions (TDM -> LDS -> WMMA)
    {
        int blocks = (SEQ / 16) * (NP1 / 128);        // 128*5 = 640
        wmma_gemm_tdm<<<blocks, 256, gemm_smem, stream>>>(emb16, WfT, bf, xpf,
                                                          SEQ, G_DIM, NP1, KP, G_DIM);
        wmma_gemm_tdm<<<blocks, 256, gemm_smem, stream>>>(emb16, WbT, bb, xpb,
                                                          SEQ, G_DIM, NP1, KP, G_DIM);
    }

    // K4: sequential biLSTM scan, U resident in LDS (f16)
    {
        size_t smem = (size_t)H_DIM * G_DIM * 2 + (G_DIM + 152) * 4; // 183008 B
        lstm_scan<<<2, 640, smem, stream>>>(xpf, xpb, Uf, Ub, hidden, hidden16);
    }

    // K5: hidden-state projections (TDM -> LDS -> WMMA)
    {
        int blocks = (SEQ / 16) * (NP2 / 128);        // 128*3 = 384
        wmma_gemm_tdm<<<blocks, 256, gemm_smem, stream>>>(hidden16, WpsT, b_ps, out_s,
                                                          SEQ, D_DIM, NP2, KP, D_DIM);
        wmma_gemm_tdm<<<blocks, 256, gemm_smem, stream>>>(hidden16, WpeT, b_pe, out_e,
                                                          SEQ, D_DIM, NP2, KP, D_DIM);
    }

    // K6: primary attention + secondary coefficients
    attention<<<SEQ, 256, 0, stream>>>(out_s, hidden, sentence, syn_idx, emb_mat,
                                       w_ss, b_ss, h_hats, coeff_s);
    attention<<<SEQ, 256, 0, stream>>>(out_e, hidden, sentence, syn_idx, emb_mat,
                                       w_se, b_se, h_bars, coeff_e);

    // K7: secondary-attention weighted sums
    weighted_sum<<<3, 256, 0, stream>>>(h_hats, coeff_s, H_HAT);
    weighted_sum<<<3, 256, 0, stream>>>(h_bars, coeff_e, H_BAR);

    // K8: logits -> d_out (8 emotion + 1 sentiment)
    final_out<<<1, 32, 0, stream>>>(H_HAT, H_BAR, W_eo, b_eo, W_so, b_so,
                                    (float*)d_out);
}
